// MoELayer_60026462929319
// MI455X (gfx1250) — compile-verified, matrix operand order checked by reading, not run on hardware
//
#include <hip/hip_runtime.h>
#include <hip/hip_bf16.h>

// ---- problem constants (from reference) ----
constexpr int kS   = 2048;
constexpr int kB   = 4;
constexpr int kD   = 1024;
constexpr int kDFF = 1408;
constexpr int kE   = 64;
constexpr int kK   = 2;
constexpr int kCAP = 1024;
constexpr int kT   = kS * kB;      // 8192 tokens
constexpr int kMB  = 64;           // M-rows per block (4 wmma tiles) for B reuse

typedef __attribute__((ext_vector_type(16))) __bf16 v16bf;
typedef __attribute__((ext_vector_type(8)))  float  v8f;
typedef unsigned int       u32;
typedef unsigned long long u64;

// ------------------------------------------------------------------
// helpers
// ------------------------------------------------------------------
__device__ __forceinline__ u64 shfl_xor_u64(u64 v, int m) {
  u32 lo = (u32)v, hi = (u32)(v >> 32);
  lo = __shfl_xor(lo, m, 32);
  hi = __shfl_xor(hi, m, 32);
  return ((u64)hi << 32) | lo;
}

// ------------------------------------------------------------------
// 0) zero output + gate accumulators (output is atomically combined)
// ------------------------------------------------------------------
__global__ void zero_kernel(float* __restrict__ y, size_t n,
                            float* __restrict__ me, int* __restrict__ counts) {
  size_t i = (size_t)blockIdx.x * 256u + threadIdx.x;
  if (i < n) y[i] = 0.0f;
  if (i < (size_t)kE) { me[i] = 0.0f; counts[i] = 0; }
}

// ------------------------------------------------------------------
// 1) tokens: x[S,B,D] fp32 -> xbf[T,D] bf16 (token-major, t = b*S+s)
// ------------------------------------------------------------------
__global__ void cvt_x_kernel(const float* __restrict__ x, __bf16* __restrict__ xbf) {
  size_t i = (size_t)blockIdx.x * 256u + threadIdx.x;
  if (i >= (size_t)kT * kD) return;
  int t = (int)(i / kD), d = (int)(i % kD);
  int b = t / kS, s = t % kS;
  xbf[i] = (__bf16)x[((size_t)s * kB + b) * kD + d];
}

// ------------------------------------------------------------------
// 2) weights: src [E][R][C] fp32 -> dst [E][C][R] bf16 (transposed so
//    each WMMA B-column is a contiguous K-run)
// ------------------------------------------------------------------
__global__ void cvt_w_kernel(const float* __restrict__ w, __bf16* __restrict__ wt,
                             int R, int C) {
  size_t i = (size_t)blockIdx.x * 256u + threadIdx.x;
  size_t rc = (size_t)R * C;
  size_t total = (size_t)kE * rc;
  if (i >= total) return;
  int e = (int)(i / rc);
  size_t rem = i % rc;
  int c = (int)(rem / R), r = (int)(rem % R);
  wt[i] = (__bf16)w[(size_t)e * rc + (size_t)r * C + c];
}

// ------------------------------------------------------------------
// 3) gate: logits -> softmax -> top-2 -> renorm weights, slot assign,
//    mean-prob accumulation for aux loss. One wave32 per token.
// ------------------------------------------------------------------
__global__ __launch_bounds__(256) void gate_kernel(
    const float* __restrict__ x, const float* __restrict__ Wg,
    int* __restrict__ rows, float* __restrict__ gatew,
    int* __restrict__ counts, float* __restrict__ me) {
  const int wave = threadIdx.x >> 5;
  const int lane = threadIdx.x & 31;
  const int t = blockIdx.x * 8 + wave;
  const int b = t / kS, s = t % kS;
  const float* xt = x + ((size_t)s * kB + b) * kD;

  // each lane owns experts {lane, lane+32}
  float acc0 = 0.0f, acc1 = 0.0f;
  for (int d = 0; d < kD; ++d) {
    float xv = xt[d];  // uniform address across lanes -> broadcast
    acc0 = fmaf(xv, Wg[d * kE + lane], acc0);
    acc1 = fmaf(xv, Wg[d * kE + lane + 32], acc1);
  }

  // softmax over the 64 logits distributed across the wave
  float m = fmaxf(acc0, acc1);
  for (int o = 16; o > 0; o >>= 1) m = fmaxf(m, __shfl_xor(m, o, 32));
  float e0 = __expf(acc0 - m), e1 = __expf(acc1 - m);
  float sum = e0 + e1;
  for (int o = 16; o > 0; o >>= 1) sum += __shfl_xor(sum, o, 32);
  float p0 = e0 / sum, p1 = e1 / sum;

  atomicAdd(&me[lane], p0 * (1.0f / (float)kT));
  atomicAdd(&me[lane + 32], p1 * (1.0f / (float)kT));

  // top-2 via packed (prob_bits << 8 | expert) max reductions
  u64 key0 = ((u64)__float_as_uint(p0) << 8) | (u32)lane;
  u64 key1 = ((u64)__float_as_uint(p1) << 8) | (u32)(lane + 32);
  u64 km = key0 > key1 ? key0 : key1;
  for (int o = 16; o > 0; o >>= 1) { u64 v = shfl_xor_u64(km, o); if (v > km) km = v; }
  const int   e_top = (int)(km & 0xFF);
  const float p_top = __uint_as_float((u32)(km >> 8));

  u64 k0b = (lane == e_top) ? 0ull : key0;
  u64 k1b = ((lane + 32) == e_top) ? 0ull : key1;
  u64 km2 = k0b > k1b ? k0b : k1b;
  for (int o = 16; o > 0; o >>= 1) { u64 v = shfl_xor_u64(km2, o); if (v > km2) km2 = v; }
  const int   e_sec = (int)(km2 & 0xFF);
  const float p_sec = __uint_as_float((u32)(km2 >> 8));

  const float inv = 1.0f / (p_top + p_sec);
  if (lane == 0) {
    int s1 = atomicAdd(&counts[e_top], 1);
    if (s1 < kCAP) { rows[e_top * kCAP + s1] = t; gatew[e_top * kCAP + s1] = p_top * inv; }
    int s2 = atomicAdd(&counts[e_sec], 1);
    if (s2 < kCAP) { rows[e_sec * kCAP + s2] = t; gatew[e_sec * kCAP + s2] = p_sec * inv; }
  }
}

// ------------------------------------------------------------------
// 4) GEMM1: h = silu(X W1) * (X W3) per expert, bf16 in / f32 acc.
//    Block = 8 waves; 64-row A panel staged in dynamic LDS and shared
//    by all waves; each wave owns a 16-col N tile and 4 M sub-tiles,
//    so every B fragment load feeds 4 WMMAs (4x weight-traffic reuse).
// ------------------------------------------------------------------
__global__ __launch_bounds__(256) void gemm1_kernel(
    const __bf16* __restrict__ xbf, const __bf16* __restrict__ w1t,
    const __bf16* __restrict__ w3t, const int* __restrict__ rows,
    const int* __restrict__ counts, __bf16* __restrict__ h) {
  const int e = blockIdx.z;
  int cnt = counts[e]; cnt = cnt > kCAP ? kCAP : cnt;
  const int mbase = blockIdx.y * kMB;
  if (mbase >= cnt) return;  // prune panels beyond routed load

  extern __shared__ __bf16 As[];  // [kMB][kD] = 128 KB (CDNA5: 320 KB/WGP)
  const int tid = threadIdx.x;
  {
    const int row = tid >> 2, q = tid & 3;   // 4 threads per row, 256 bf16 each
    const int r = mbase + row;
    const int tok = (r < cnt) ? rows[e * kCAP + r] : 0;
    const uint4* src = (const uint4*)(xbf + (size_t)tok * kD + q * 256);
    uint4* dst = (uint4*)(As + (size_t)row * kD + q * 256);
#pragma unroll
    for (int i = 0; i < 16; ++i) dst[i] = src[i];
  }
  __syncthreads();

  const int wave = tid >> 5, lane = tid & 31;
  const int ncol = blockIdx.x * 128 + wave * 16 + (lane & 15);
  const int koff = (lane < 16) ? 0 : 8;   // ISA 16-bit operand lane layout
  const int arow = lane & 15;
  const __bf16* b1p = w1t + ((size_t)e * kDFF + ncol) * kD;
  const __bf16* b3p = w3t + ((size_t)e * kDFF + ncol) * kD;

  v8f c1[4] = {}, c3[4] = {};
  for (int kk = 0; kk < kD; kk += 32) {
    __builtin_prefetch(b1p + kk + 64, 0, 1);
    __builtin_prefetch(b3p + kk + 64, 0, 1);
    v16bf b1, b3;
    ((uint4*)&b1)[0] = *(const uint4*)(b1p + kk + koff);
    ((uint4*)&b1)[1] = *(const uint4*)(b1p + kk + koff + 16);
    ((uint4*)&b3)[0] = *(const uint4*)(b3p + kk + koff);
    ((uint4*)&b3)[1] = *(const uint4*)(b3p + kk + koff + 16);
#pragma unroll
    for (int mg = 0; mg < 4; ++mg) {
      v16bf a;
      const __bf16* ap = As + (size_t)(mg * 16 + arow) * kD + kk + koff;
      ((uint4*)&a)[0] = *(const uint4*)ap;
      ((uint4*)&a)[1] = *(const uint4*)(ap + 16);
      c1[mg] = __builtin_amdgcn_wmma_f32_16x16x32_bf16(false, a, false, b1, (short)0, c1[mg], false, false);
      c3[mg] = __builtin_amdgcn_wmma_f32_16x16x32_bf16(false, a, false, b3, (short)0, c3[mg], false, false);
    }
  }

  __bf16* hp = h + (size_t)e * kCAP * kDFF + ncol;
#pragma unroll
  for (int mg = 0; mg < 4; ++mg) {
    const int rb = mbase + mg * 16 + ((lane < 16) ? 0 : 8);
#pragma unroll
    for (int v = 0; v < 8; ++v) {
      float g = c1[mg][v], u = c3[mg][v];
      float hv = (g / (1.0f + __expf(-g))) * u;  // silu(g) * u
      hp[(size_t)(rb + v) * kDFF] = (__bf16)hv;
    }
  }
}

// ------------------------------------------------------------------
// 5) GEMM2: out = h W2 per expert (same 64-row panel blocking), then
//    weighted atomic combine into y.
// ------------------------------------------------------------------
__global__ __launch_bounds__(256) void gemm2_kernel(
    const __bf16* __restrict__ h, const __bf16* __restrict__ w2t,
    const int* __restrict__ rows, const float* __restrict__ gatew,
    const int* __restrict__ counts, float* __restrict__ y) {
  const int e = blockIdx.z;
  int cnt = counts[e]; cnt = cnt > kCAP ? kCAP : cnt;
  const int mbase = blockIdx.y * kMB;
  if (mbase >= cnt) return;

  extern __shared__ __bf16 As[];  // [kMB][kDFF] = 176 KB
  const int tid = threadIdx.x;
  {
    const int row = tid >> 2, q = tid & 3;   // 4 threads per row, 352 bf16 each
    const uint4* src = (const uint4*)(h + ((size_t)e * kCAP + mbase + row) * kDFF + q * 352);
    uint4* dst = (uint4*)(As + (size_t)row * kDFF + q * 352);
#pragma unroll
    for (int i = 0; i < 22; ++i) dst[i] = src[i];
  }
  __syncthreads();

  const int wave = tid >> 5, lane = tid & 31;
  const int ncol = blockIdx.x * 128 + wave * 16 + (lane & 15);
  const int koff = (lane < 16) ? 0 : 8;
  const int arow = lane & 15;
  const __bf16* bp = w2t + ((size_t)e * kD + ncol) * kDFF;

  v8f c[4] = {};
  for (int kk = 0; kk < kDFF; kk += 32) {
    __builtin_prefetch(bp + kk + 64, 0, 1);
    v16bf b;
    ((uint4*)&b)[0] = *(const uint4*)(bp + kk + koff);
    ((uint4*)&b)[1] = *(const uint4*)(bp + kk + koff + 16);
#pragma unroll
    for (int mg = 0; mg < 4; ++mg) {
      v16bf a;
      const __bf16* ap = As + (size_t)(mg * 16 + arow) * kDFF + kk + koff;
      ((uint4*)&a)[0] = *(const uint4*)ap;
      ((uint4*)&a)[1] = *(const uint4*)(ap + 16);
      c[mg] = __builtin_amdgcn_wmma_f32_16x16x32_bf16(false, a, false, b, (short)0, c[mg], false, false);
    }
  }

#pragma unroll
  for (int mg = 0; mg < 4; ++mg) {
    const int rb = mbase + mg * 16 + ((lane < 16) ? 0 : 8);
#pragma unroll
    for (int v = 0; v < 8; ++v) {
      const int r = rb + v;
      if (r < cnt) {
        const int tok = rows[e * kCAP + r];
        const float gw = gatew[e * kCAP + r];
        const int bb = tok / kS, ss = tok % kS;
        atomicAdd(&y[((size_t)ss * kB + bb) * kD + ncol], gw * c[mg][v]);
      }
    }
  }
}

// ------------------------------------------------------------------
// 6) aux loss: l_aux = E * sum_e me[e] * (count[e] / (T*K))
// ------------------------------------------------------------------
__global__ void laux_kernel(const float* __restrict__ me, const int* __restrict__ counts,
                            float* __restrict__ out) {
  __shared__ float p[kE];
  int e = threadIdx.x;
  p[e] = me[e] * ((float)counts[e] / (float)(kT * kK));
  __syncthreads();
  if (e == 0) {
    float s = 0.0f;
    for (int i = 0; i < kE; ++i) s += p[i];
    out[0] = (float)kE * s;
  }
}

// ------------------------------------------------------------------
extern "C" void kernel_launch(void* const* d_in, const int* in_sizes, int n_in,
                              void* d_out, int out_size, void* d_ws, size_t ws_size,
                              hipStream_t stream) {
  (void)in_sizes; (void)n_in; (void)out_size; (void)ws_size;
  const float* x  = (const float*)d_in[0];  // [S,B,D]
  const float* Wg = (const float*)d_in[1];  // [D,E]
  const float* W1 = (const float*)d_in[2];  // [E,D,DFF]
  const float* W3 = (const float*)d_in[3];  // [E,D,DFF]
  const float* W2 = (const float*)d_in[4];  // [E,DFF,D]
  float* y    = (float*)d_out;              // [S,B,D] then l_aux
  float* laux = y + (size_t)kT * kD;

  char* ws = (char*)d_ws;
  size_t off = 0;
  const size_t WEL = (size_t)kE * kD * kDFF;
  __bf16* w1t  = (__bf16*)(ws + off); off += WEL * 2;                 // [E][DFF][D]
  __bf16* w3t  = (__bf16*)(ws + off); off += WEL * 2;                 // [E][DFF][D]
  __bf16* w2t  = (__bf16*)(ws + off); off += WEL * 2;                 // [E][D][DFF]
  __bf16* xbf  = (__bf16*)(ws + off); off += (size_t)kT * kD * 2;     // [T][D]
  __bf16* hbuf = (__bf16*)(ws + off); off += (size_t)kE * kCAP * kDFF * 2;
  int*    rows = (int*)  (ws + off);  off += (size_t)kE * kCAP * 4;
  float* gatew = (float*)(ws + off);  off += (size_t)kE * kCAP * 4;
  int*  counts = (int*)  (ws + off);  off += kE * 4;
  float*    me = (float*)(ws + off);  off += kE * 4;

  const size_t nY = (size_t)kT * kD;
  zero_kernel <<<(unsigned)((nY + 255) / 256), 256, 0, stream>>>(y, nY, me, counts);
  cvt_x_kernel<<<(unsigned)((nY + 255) / 256), 256, 0, stream>>>(x, xbf);
  cvt_w_kernel<<<(unsigned)((WEL + 255) / 256), 256, 0, stream>>>(W1, w1t, kD, kDFF);
  cvt_w_kernel<<<(unsigned)((WEL + 255) / 256), 256, 0, stream>>>(W3, w3t, kD, kDFF);
  cvt_w_kernel<<<(unsigned)((WEL + 255) / 256), 256, 0, stream>>>(W2, w2t, kDFF, kD);
  gate_kernel <<<kT / 8, 256, 0, stream>>>(x, Wg, rows, gatew, counts, me);

  const unsigned lds1 = (unsigned)(kMB * kD * 2);    // 128 KB A-panel
  const unsigned lds2 = (unsigned)(kMB * kDFF * 2);  // 176 KB A-panel
  gemm1_kernel<<<dim3(kDFF / 128, kCAP / kMB, kE), 256, lds1, stream>>>(xbf, w1t, w3t, rows, counts, hbuf);
  gemm2_kernel<<<dim3(kD / 128, kCAP / kMB, kE), 256, lds2, stream>>>(hbuf, w2t, rows, gatew, counts, y);
  laux_kernel <<<1, kE, 0, stream>>>(me, counts, laux);
}
